// GraphRecLSTM_87471303950596
// MI455X (gfx1250) — compile-verified, matrix operand order checked by reading, not run on hardware
//
#include <hip/hip_runtime.h>
#include <hip/hip_bf16.h>

// ---------------------------------------------------------------------------
// GraphRec forward on gfx1250: all GEMMs via v_wmma_f32_16x16x32_f16 (wave32),
// LSTM seq tiles staged via the Tensor Data Mover (TENSOR_LOAD_TO_LDS).
// ---------------------------------------------------------------------------

typedef __attribute__((ext_vector_type(16))) _Float16 v16h;
typedef __attribute__((ext_vector_type(8)))  float    v8f;
typedef __attribute__((ext_vector_type(4)))  float    f32x4;

constexpr int BATCH = 8192;
constexpr int SEQL  = 50;
constexpr int EDIM  = 128;   // E
constexpr int HDIM  = 128;   // H
constexpr int GDIM  = 512;   // 4H
constexpr int ROWS  = 32;    // batch rows per LSTM block

#if defined(__HIP_DEVICE_COMPILE__) && __has_builtin(__builtin_amdgcn_tensor_load_to_lds)
#define HAVE_TDM 1
#else
#define HAVE_TDM 0
#endif

__device__ __forceinline__ float sigm(float x) { return 1.f / (1.f + __expf(-x)); }
__device__ __forceinline__ float tanh_fast(float x) { return 2.f * sigm(2.f * x) - 1.f; }

__device__ __forceinline__ v8f wmma_f16(v16h a, v16h b, v8f c) {
  return __builtin_amdgcn_wmma_f32_16x16x32_f16(false, a, false, b, (short)0, c,
                                                false, false);
}

// Load one 16x32 f16 operand tile (A: MxK, or B as W^T where W is [N,K] row-
// major). Per ISA 7.12.2: lane&15 selects the row; lane>>4 selects which K
// half-group; each lane holds K pairs {kh..kh+7, 16+kh..16+kh+7}.
__device__ __forceinline__ v16h load_ab_tile(const _Float16* __restrict__ base, int ld) {
  const int lane = threadIdx.x & 31;
  const int r    = lane & 15;
  const int kh   = (lane >> 4) << 3;
  const _Float16* p = base + (size_t)r * ld + kh;
  union { v16h v; struct { f32x4 lo, hi; } q; } u;
  u.q.lo = *(const f32x4*)(p);
  u.q.hi = *(const f32x4*)(p + 16);
  return u.v;
}

#if HAVE_TDM
typedef __attribute__((ext_vector_type(4))) unsigned int u32x4;
typedef __attribute__((ext_vector_type(8))) int          i32x8;
typedef __attribute__((ext_vector_type(4))) int          i32x4;

// 2-D tile DMA: rows x cols (f16), row pitch row_stride elems, global -> LDS.
// D# layout per CDNA5 ISA ch.8 (group0: count/lds/global/type, group1: dims).
// This toolchain exposes the 6-arg builtin: (g0, g1, g2, g3, g2b, cpol).
__device__ __forceinline__ void tdm_load_tile_f16(const _Float16* gsrc, void* lds_dst,
                                                  int rows, int cols, int row_stride) {
  unsigned long long ga = (unsigned long long)(uintptr_t)gsrc;   // byte address
  unsigned lds = (unsigned)(uintptr_t)lds_dst;                   // LDS byte offset
  u32x4 g0;
  g0[0] = 1u;                                                    // count=1 (valid)
  g0[1] = lds;                                                   // lds_addr
  g0[2] = (unsigned)(ga & 0xffffffffu);                          // global_addr lo
  g0[3] = (unsigned)((ga >> 32) & 0x01ffffffu) | (2u << 30);     // global hi | type=2
  i32x8 g1;
  g1[0] = 1 << 16;                                 // data_size=1 (2 bytes)
  g1[1] = (cols & 0xffff) << 16;                   // tensor_dim0 lo16 @ [63:48]
  g1[2] = ((unsigned)cols >> 16) | ((rows & 0xffff) << 16);  // dim0 hi | dim1 lo
  g1[3] = ((unsigned)rows >> 16) | ((cols & 0xffff) << 16);  // dim1 hi | tile_dim0
  g1[4] = rows & 0xffff;                           // tile_dim1 (tile_dim2=0)
  g1[5] = row_stride;                              // tensor_dim0_stride lo32
  g1[6] = 0;                                       // stride hi | dim1_stride lo
  g1[7] = 0;
  i32x4 z4 = {0, 0, 0, 0};
  i32x8 z8 = {0, 0, 0, 0, 0, 0, 0, 0};
  __builtin_amdgcn_tensor_load_to_lds(g0, g1, z4, z4, z8, 0);
}
#endif

// -------------------------- prep kernels -----------------------------------

__global__ void k_f32_to_f16(const float* __restrict__ src, _Float16* __restrict__ dst, int n) {
  int i = blockIdx.x * blockDim.x + threadIdx.x;
  if (i < n) dst[i] = (_Float16)src[i];
}

struct GatherArgs { const float* emb; const int* idx; _Float16* dst; };

__global__ void __launch_bounds__(128)
k_gather(GatherArgs u, GatherArgs v) {
  GatherArgs a = blockIdx.y ? v : u;
  int b = blockIdx.x, col = threadIdx.x;
  a.dst[(size_t)b * EDIM + col] = (_Float16)a.emb[(size_t)a.idx[b] * EDIM + col];
}

struct BuildArgs { const float* emb; const int* items; const int* ratings;
                   const int* lens; _Float16* seq; };

// seq[b,t,:] = emb[items[b,t]] * r2e[ratings[b,t]] * (t < len[b])
__global__ void __launch_bounds__(128)
k_build_seq(BuildArgs u, BuildArgs v, const float* __restrict__ r2e) {
  BuildArgs a = blockIdx.y ? v : u;
  int bt = blockIdx.x;
  int b = bt / SEQL, t = bt % SEQL;
  int col = threadIdx.x;
  int it = a.items[bt], rt = a.ratings[bt];
  float val = (t < a.lens[b])
                  ? a.emb[(size_t)it * EDIM + col] * r2e[(size_t)rt * EDIM + col]
                  : 0.f;
  a.seq[(size_t)bt * EDIM + col] = (_Float16)val;
}

// -------------------------- fused LSTM scan --------------------------------
// g = seq_t @ Wih^T + h @ Whh^T + (bih+bhh) per step (x_pre fused into scan),
// gate nonlinearity in VALU, h double-buffered in LDS, c in registers, seq
// tiles DMA'd one step ahead by the TDM. blockIdx.y selects user/item side.
struct ScanArgs {
  const _Float16* seq;   // [B,L,E] f16
  const _Float16* wih;   // [4H,E]  f16
  const _Float16* whh;   // [4H,H]  f16
  const float* bih; const float* bhh;
  const int* len;
  _Float16* outh;        // [B,L,H] f16 (length-masked)
};

__global__ void __launch_bounds__(256)
k_lstm_scan(ScanArgs ua, ScanArgs va) {
  const ScanArgs a = blockIdx.y ? va : ua;

  __shared__ _Float16 sX[2][ROWS * EDIM];      // 16 KB (double-buffered DMA dst)
  __shared__ _Float16 sH[2][ROWS * HDIM];      // 16 KB
  __shared__ float    sBias[GDIM];             // 2 KB
  __shared__ int      sLen[ROWS];

  const int rowBase = blockIdx.x * ROWS;
  const int tid  = threadIdx.x;
  const int wave = tid >> 5;
  const int lane = tid & 31;
  constexpr int MT = ROWS / 16;

  for (int i = tid; i < GDIM; i += 256) sBias[i] = a.bih[i] + a.bhh[i];
  for (int i = tid; i < ROWS; i += 256) sLen[i] = a.len[rowBase + i];
  for (int i = tid; i < ROWS * HDIM; i += 256) sH[0][i] = (_Float16)0.f;
  __syncthreads();

#if HAVE_TDM
  if (tid == 0)
    tdm_load_tile_f16(a.seq + (size_t)rowBase * SEQL * EDIM, sX[0], ROWS, EDIM,
                      SEQL * EDIM);
#endif

  const int ncol = wave * 16 + (lane & 15);
  const float bi = sBias[0 * HDIM + ncol];
  const float bf = sBias[1 * HDIM + ncol];
  const float bg = sBias[2 * HDIM + ncol];
  const float bo = sBias[3 * HDIM + ncol];

  v8f cst[MT];
  v8f vzero = {};
  for (int mt = 0; mt < MT; ++mt) cst[mt] = vzero;

  int cur = 0, curX = 0;
  for (int t = 0; t < SEQL; ++t) {
#if HAVE_TDM
    if (tid == 0) __builtin_amdgcn_s_wait_tensorcnt(0);
#else
    for (int i = tid; i < ROWS * EDIM / 8; i += 256) {
      int m = i >> 4;
      int kc = (i & 15) << 3;
      *(f32x4*)(sX[curX] + m * EDIM + kc) =
          *(const f32x4*)(a.seq + ((size_t)(rowBase + m) * SEQL + t) * EDIM + kc);
    }
#endif
    __syncthreads();  // sX[curX] ready; previous step fully done (sX[curX^1] free)
#if HAVE_TDM
    if (tid == 0 && t + 1 < SEQL)
      tdm_load_tile_f16(a.seq + ((size_t)rowBase * SEQL + (t + 1)) * EDIM,
                        sX[curX ^ 1], ROWS, EDIM, SEQL * EDIM);
#endif
    const int nxt = cur ^ 1;

    for (int mt = 0; mt < MT; ++mt) {
      v16h aX[4], aH[4];
#pragma unroll
      for (int kt = 0; kt < 4; ++kt)
        aX[kt] = load_ab_tile(sX[curX] + (mt * 16) * EDIM + kt * 32, EDIM);
#pragma unroll
      for (int kt = 0; kt < 4; ++kt)
        aH[kt] = load_ab_tile(sH[cur] + (mt * 16) * HDIM + kt * 32, HDIM);

      v8f acc[4];
#pragma unroll
      for (int g = 0; g < 4; ++g) {
        const int nt = g * 8 + wave;
        v8f s = vzero;
#pragma unroll
        for (int kt = 0; kt < 4; ++kt)
          s = wmma_f16(aX[kt], load_ab_tile(a.wih + (size_t)(nt * 16) * EDIM + kt * 32, EDIM), s);
#pragma unroll
        for (int kt = 0; kt < 4; ++kt)
          s = wmma_f16(aH[kt], load_ab_tile(a.whh + (size_t)(nt * 16) * HDIM + kt * 32, HDIM), s);
        acc[g] = s;
      }
      // gate math; D element r -> row mt*16 + (lane>>4)*8 + r, col ncol
#pragma unroll
      for (int r = 0; r < 8; ++r) {
        float i_ = sigm(acc[0][r] + bi);
        float f_ = sigm(acc[1][r] + bf);
        float g_ = tanh_fast(acc[2][r] + bg);
        float o_ = sigm(acc[3][r] + bo);
        float c  = f_ * cst[mt][r] + i_ * g_;
        cst[mt][r] = c;
        float h = o_ * tanh_fast(c);
        int m = mt * 16 + ((lane >> 4) << 3) + r;
        sH[nxt][m * HDIM + ncol] = (_Float16)h;
      }
    }
    __syncthreads();
    // masked, coalesced writeback of this step's hidden state
    for (int i = tid; i < ROWS * HDIM / 8; i += 256) {
      int m = i >> 4;
      int kc = (i & 15) << 3;
      f32x4 v = *(const f32x4*)(sH[nxt] + m * HDIM + kc);
      f32x4 z = {};
      if (t >= sLen[m]) v = z;
      *(f32x4*)(a.outh + ((size_t)(rowBase + m) * SEQL + t) * HDIM + kc) = v;
    }
    cur = nxt;
    curX ^= 1;
  }
}

// ------------------- attention scores: tanh(out@A1^T+b)@A2 + b2 ------------
struct AttnArgs {
  const _Float16* outh;  // [B*L, H] f16
  const _Float16* a1;    // [H,H] f16
  const float* a1b; const float* a2; const float* a2b;
  float* scores;         // [B*L]
};

__global__ void __launch_bounds__(128)
k_attn_scores(AttnArgs ua, AttnArgs va) {
  const AttnArgs a = blockIdx.y ? va : ua;
  __shared__ _Float16 sA1[HDIM * HDIM];  // 32 KB
  __shared__ float sB1[HDIM], sA2[HDIM];
  const int tid = threadIdx.x, wave = tid >> 5, lane = tid & 31;

  for (int i = tid; i < HDIM * HDIM / 8; i += 128)
    *(f32x4*)(sA1 + i * 8) = *(const f32x4*)(a.a1 + i * 8);
  for (int i = tid; i < HDIM; i += 128) { sB1[i] = a.a1b[i]; sA2[i] = a.a2[i]; }
  __syncthreads();

  const size_t rowBase = (size_t)blockIdx.x * 64 + wave * 16;
  v16h aT[4];
#pragma unroll
  for (int kt = 0; kt < 4; ++kt)
    aT[kt] = load_ab_tile(a.outh + rowBase * HDIM + kt * 32, HDIM);

  v8f sum = {};
#pragma unroll
  for (int nt = 0; nt < 8; ++nt) {
    v8f acc = {};
#pragma unroll
    for (int kt = 0; kt < 4; ++kt)
      acc = wmma_f16(aT[kt], load_ab_tile(sA1 + (nt * 16) * HDIM + kt * 32, HDIM), acc);
    int n = nt * 16 + (lane & 15);
    float b1 = sB1[n], w2 = sA2[n];
#pragma unroll
    for (int r = 0; r < 8; ++r) sum[r] += tanh_fast(acc[r] + b1) * w2;
  }
  // reduce over the 16 lanes that share a row
#pragma unroll
  for (int off = 1; off < 16; off <<= 1) {
#pragma unroll
    for (int r = 0; r < 8; ++r) sum[r] += __shfl_xor(sum[r], off, 32);
  }
  if ((lane & 15) == 0) {
    float bb = a.a2b[0];
#pragma unroll
    for (int r = 0; r < 8; ++r)
      a.scores[rowBase + (size_t)(lane >> 4) * 8 + r] = sum[r] + bb;
  }
}

// ---------------- softmax over T + weighted sum -> dst (f16) ---------------
struct PoolArgs { const float* scores; const _Float16* outh; _Float16* dst; };

__global__ void __launch_bounds__(128)
k_attn_pool(PoolArgs ua, PoolArgs va, int ldc) {
  const PoolArgs a = blockIdx.y ? va : ua;
  __shared__ float sS[SEQL];
  const int b = blockIdx.x, tid = threadIdx.x;
  for (int i = tid; i < SEQL; i += 128) sS[i] = a.scores[(size_t)b * SEQL + i];
  __syncthreads();
  float mx = -1e30f;
  for (int t = 0; t < SEQL; ++t) mx = fmaxf(mx, sS[t]);
  float den = 0.f;
  for (int t = 0; t < SEQL; ++t) den += __expf(sS[t] - mx);
  float inv = 1.f / den;
  float acc = 0.f;
  const int col = tid;
  for (int t = 0; t < SEQL; ++t) {
    float w = __expf(sS[t] - mx) * inv;
    acc += w * (float)a.outh[((size_t)b * SEQL + t) * HDIM + col];
  }
  a.dst[(size_t)b * ldc + col] = (_Float16)acc;
}

// ------- generic WMMA GEMM: out = act(bn(A @ W^T + bias)) (f16 out) --------
template <int KD, int ND>
__global__ void __launch_bounds__(128)
k_gemm_mlp(const _Float16* __restrict__ A,   // [M, KD] f16
           const _Float16* __restrict__ W,   // [ND, KD] f16
           const float* __restrict__ bias, const float* __restrict__ gamma,
           const float* __restrict__ beta, int relu,
           _Float16* __restrict__ out, int ldc, int colOff) {
  const int tid = threadIdx.x, wave = tid >> 5, lane = tid & 31;
  const size_t rowBase = (size_t)blockIdx.x * 64 + wave * 16;

  v16h aT[KD / 32];
#pragma unroll
  for (int kt = 0; kt < KD / 32; ++kt)
    aT[kt] = load_ab_tile(A + rowBase * KD + kt * 32, KD);

#pragma unroll
  for (int nt = 0; nt < ND / 16; ++nt) {
    v8f acc = {};
#pragma unroll
    for (int kt = 0; kt < KD / 32; ++kt)
      acc = wmma_f16(aT[kt], load_ab_tile(W + (size_t)(nt * 16) * KD + kt * 32, KD), acc);
    int n = nt * 16 + (lane & 15);
    float bv = bias[n];
    float s = 1.f, sh = 0.f;
    if (gamma) { s = gamma[n] * rsqrtf(1.f + 1e-5f); sh = beta[n]; }
#pragma unroll
    for (int r = 0; r < 8; ++r) {
      float v = (acc[r] + bv) * s + sh;
      if (relu) v = fmaxf(v, 0.f);
      out[(rowBase + ((lane >> 4) << 3) + r) * ldc + colOff + n] = (_Float16)v;
    }
  }
}

// ---------------------- final 16 -> 1 projection ---------------------------
__global__ void k_head(const _Float16* __restrict__ x2, const float* __restrict__ w3,
                       const float* __restrict__ b3, float* __restrict__ out, int n) {
  int i = blockIdx.x * blockDim.x + threadIdx.x;
  if (i >= n) return;
  float acc = b3[0];
#pragma unroll
  for (int j = 0; j < 16; ++j) acc += (float)x2[(size_t)i * 16 + j] * w3[j];
  out[i] = acc;
}

// ---------------------------------------------------------------------------

extern "C" void kernel_launch(void* const* d_in, const int* in_sizes, int n_in,
                              void* d_out, int out_size, void* d_ws, size_t ws_size,
                              hipStream_t stream) {
  (void)in_sizes; (void)n_in; (void)out_size; (void)ws_size;

  // ---- inputs (setup_inputs dict order; params dict flattened in order) ----
  const int* nodes_u = (const int*)d_in[0];
  const int* nodes_v = (const int*)d_in[1];
  const int* u_items = (const int*)d_in[2];
  const int* u_rat   = (const int*)d_in[3];
  const int* u_len   = (const int*)d_in[4];
  const int* i_users = (const int*)d_in[5];
  const int* i_rat   = (const int*)d_in[6];
  const int* i_len   = (const int*)d_in[7];
  const float* u2e = (const float*)d_in[8];
  const float* v2e = (const float*)d_in[9];
  const float* r2e = (const float*)d_in[10];
  // user_agg: Wih(11) Whh(12) bih(13) bhh(14) A1(15) a1b(16) A2(17) a2b(18)
  // item_agg: 19..26 same order
  const float* bn1_g = (const float*)d_in[27];
  const float* bn1_b = (const float*)d_in[28];
  const float* bn2_g = (const float*)d_in[29];
  const float* bn2_b = (const float*)d_in[30];
  const float* bn3_g = (const float*)d_in[31];
  const float* bn3_b = (const float*)d_in[32];
  const float* bn4_g = (const float*)d_in[33];
  const float* bn4_b = (const float*)d_in[34];
  // 35..48: w_ur1_W/b, w_ur2_W/b, w_vr1_W/b, w_vr2_W/b, w_uv1_W/b, w_uv2_W/b, w_uv3_W/b

  // ---- workspace carve-out ----
  char* p = (char*)d_ws;
  auto alloc = [&](size_t bytes) -> void* {
    void* r = (void*)p;
    p += (bytes + 255) & ~(size_t)255;
    return r;
  };
  _Float16* seqU  = (_Float16*)alloc((size_t)BATCH * SEQL * EDIM * 2);
  _Float16* seqI  = (_Float16*)alloc((size_t)BATCH * SEQL * EDIM * 2);
  _Float16* outU  = (_Float16*)alloc((size_t)BATCH * SEQL * HDIM * 2);
  _Float16* outI  = (_Float16*)alloc((size_t)BATCH * SEQL * HDIM * 2);
  _Float16* wihU  = (_Float16*)alloc((size_t)GDIM * EDIM * 2);
  _Float16* whhU  = (_Float16*)alloc((size_t)GDIM * HDIM * 2);
  _Float16* a1U   = (_Float16*)alloc((size_t)HDIM * HDIM * 2);
  _Float16* wihI  = (_Float16*)alloc((size_t)GDIM * EDIM * 2);
  _Float16* whhI  = (_Float16*)alloc((size_t)GDIM * HDIM * 2);
  _Float16* a1I   = (_Float16*)alloc((size_t)HDIM * HDIM * 2);
  _Float16* wur1  = (_Float16*)alloc((size_t)EDIM * EDIM * 2);
  _Float16* wur2  = (_Float16*)alloc((size_t)EDIM * EDIM * 2);
  _Float16* wvr1  = (_Float16*)alloc((size_t)EDIM * EDIM * 2);
  _Float16* wvr2  = (_Float16*)alloc((size_t)EDIM * EDIM * 2);
  _Float16* wuv1  = (_Float16*)alloc((size_t)EDIM * 512 * 2);
  _Float16* wuv2  = (_Float16*)alloc((size_t)16 * EDIM * 2);
  float*    scU   = (float*)alloc((size_t)BATCH * SEQL * 4);
  float*    scI   = (float*)alloc((size_t)BATCH * SEQL * 4);
  _Float16* eU    = (_Float16*)alloc((size_t)BATCH * EDIM * 2);
  _Float16* eI    = (_Float16*)alloc((size_t)BATCH * EDIM * 2);
  _Float16* xu1   = (_Float16*)alloc((size_t)BATCH * EDIM * 2);
  _Float16* xv1   = (_Float16*)alloc((size_t)BATCH * EDIM * 2);
  _Float16* x_uv  = (_Float16*)alloc((size_t)BATCH * 512 * 2);
  _Float16* x1    = (_Float16*)alloc((size_t)BATCH * EDIM * 2);
  _Float16* x2    = (_Float16*)alloc((size_t)BATCH * 16 * 2);

  auto cvt = [&](int idx, _Float16* dst, int n) {
    k_f32_to_f16<<<(n + 255) / 256, 256, 0, stream>>>((const float*)d_in[idx], dst, n);
  };
  // weight conversions
  cvt(11, wihU, GDIM * EDIM);
  cvt(12, whhU, GDIM * HDIM);
  cvt(15, a1U,  HDIM * HDIM);
  cvt(19, wihI, GDIM * EDIM);
  cvt(20, whhI, GDIM * HDIM);
  cvt(23, a1I,  HDIM * HDIM);
  cvt(35, wur1, EDIM * EDIM);
  cvt(37, wur2, EDIM * EDIM);
  cvt(39, wvr1, EDIM * EDIM);
  cvt(41, wvr2, EDIM * EDIM);
  cvt(43, wuv1, EDIM * 512);
  cvt(45, wuv2, 16 * EDIM);

  // gathers + masked sequence build (user/item fused via blockIdx.y)
  k_gather<<<dim3(BATCH, 2), 128, 0, stream>>>(
      GatherArgs{u2e, nodes_u, eU}, GatherArgs{v2e, nodes_v, eI});
  k_build_seq<<<dim3(BATCH * SEQL, 2), 128, 0, stream>>>(
      BuildArgs{v2e, u_items, u_rat, u_len, seqU},
      BuildArgs{u2e, i_users, i_rat, i_len, seqI}, r2e);

  // fused LSTM scans (both sides in one launch)
  k_lstm_scan<<<dim3(BATCH / ROWS, 2), 256, 0, stream>>>(
      ScanArgs{seqU, wihU, whhU, (const float*)d_in[13], (const float*)d_in[14],
               u_len, outU},
      ScanArgs{seqI, wihI, whhI, (const float*)d_in[21], (const float*)d_in[22],
               i_len, outI});

  // attention scores + pooling (pool writes straight into x_uv columns)
  k_attn_scores<<<dim3(BATCH * SEQL / 64, 2), 128, 0, stream>>>(
      AttnArgs{outU, a1U, (const float*)d_in[16], (const float*)d_in[17],
               (const float*)d_in[18], scU},
      AttnArgs{outI, a1I, (const float*)d_in[24], (const float*)d_in[25],
               (const float*)d_in[26], scI});
  k_attn_pool<<<dim3(BATCH, 2), 128, 0, stream>>>(
      PoolArgs{scU, outU, x_uv + 256}, PoolArgs{scI, outI, x_uv + 384}, 512);

  // MLP head
  const int gblk = BATCH / 64;
  k_gemm_mlp<128, 128><<<gblk, 128, 0, stream>>>(
      eU, wur1, (const float*)d_in[36], bn1_g, bn1_b, 1, xu1, 128, 0);
  k_gemm_mlp<128, 128><<<gblk, 128, 0, stream>>>(
      xu1, wur2, (const float*)d_in[38], nullptr, nullptr, 0, x_uv, 512, 0);
  k_gemm_mlp<128, 128><<<gblk, 128, 0, stream>>>(
      eI, wvr1, (const float*)d_in[40], bn2_g, bn2_b, 1, xv1, 128, 0);
  k_gemm_mlp<128, 128><<<gblk, 128, 0, stream>>>(
      xv1, wvr2, (const float*)d_in[42], nullptr, nullptr, 0, x_uv, 512, 128);
  k_gemm_mlp<512, 128><<<gblk, 128, 0, stream>>>(
      x_uv, wuv1, (const float*)d_in[44], bn3_g, bn3_b, 1, x1, 128, 0);
  k_gemm_mlp<128, 16><<<gblk, 128, 0, stream>>>(
      x1, wuv2, (const float*)d_in[46], bn4_g, bn4_b, 1, x2, 16, 0);
  k_head<<<(BATCH + 255) / 256, 256, 0, stream>>>(
      x2, (const float*)d_in[47], (const float*)d_in[48], (float*)d_out, BATCH);
}
